// Mamba_70746701300346
// MI455X (gfx1250) — compile-verified, matrix operand order checked by reading, not run on hardware
//
#include <hip/hip_runtime.h>
#include <hip/hip_bf16.h>
#include <math.h>

#define B__ 2
#define L__ 2048
#define DM_ 1024
#define DI_ 2048
#define NS_ 16
#define RR_ 64
#define KC_ 4
#define ML_ (B__ * L__)          // 4096 GEMM rows
#define XPN (RR_ + 2 * NS_)      // 96
#define XPNP 128                 // x_proj N padded to multiple of 64

typedef __attribute__((ext_vector_type(16))) __bf16        v16bf;
typedef __attribute__((ext_vector_type(8)))  float         v8f;
typedef __attribute__((ext_vector_type(8)))  unsigned int  v8u;
typedef __attribute__((ext_vector_type(4)))  unsigned int  u32x4;
typedef __attribute__((ext_vector_type(8)))  int           i32x8;
typedef __attribute__((ext_vector_type(4)))  int           i32x4;

#if __has_builtin(__builtin_amdgcn_tensor_load_to_lds) && __has_builtin(__builtin_amdgcn_s_wait_tensorcnt)
#define HAVE_TDM 1
#else
#define HAVE_TDM 0
#endif

// ---- TDM tile geometry: 64 rows x 128 bf16 per chunk, 272B padded LDS rows
#define KCH     128
#define ROWU    136              // ushorts per LDS row (256B data + 16B pad)
#define LDSBUFB (64 * ROWU * 2)  // 17408 bytes per buffer

__device__ __forceinline__ unsigned short f2bf(float f) {
  unsigned int u = __float_as_uint(f);
  unsigned int r = u + 0x7FFFu + ((u >> 16) & 1u);   // round-to-nearest-even
  return (unsigned short)(r >> 16);
}
__device__ __forceinline__ float sigmoidf_(float x) { return 1.0f / (1.0f + __expf(-x)); }

// ---------------- elementwise cast fp32 -> bf16 ----------------
__global__ void cast_f32_bf16_k(const float* __restrict__ src,
                                unsigned short* __restrict__ dst, int n) {
  int i = blockIdx.x * blockDim.x + threadIdx.x;
  if (i < n) dst[i] = f2bf(src[i]);
}

// cast with zero-padded extra rows: dst has rows_pad rows of K each
__global__ void cast_pad_bf16_k(const float* __restrict__ src,
                                unsigned short* __restrict__ dst,
                                int rows, int rows_pad, int K) {
  int i = blockIdx.x * blockDim.x + threadIdx.x;
  if (i >= rows_pad * K) return;
  int row = i / K;
  dst[i] = (row < rows) ? f2bf(src[i]) : (unsigned short)0;
}

// take first `cols` of each row (row stride srs) -> packed bf16
__global__ void slice_bf16_k(const float* __restrict__ src,
                             unsigned short* __restrict__ dst,
                             int rows, int cols, int srs) {
  int i = blockIdx.x * blockDim.x + threadIdx.x;
  if (i >= rows * cols) return;
  int row = i / cols, col = i - row * cols;
  dst[i] = f2bf(src[(size_t)row * srs + col]);
}

// ---------------- causal depthwise conv (K=4) + SiLU ----------------
__global__ void conv_silu_k(const float* __restrict__ xz,
                            const float* __restrict__ cw,
                            const float* __restrict__ cb,
                            float* __restrict__ xact,
                            unsigned short* __restrict__ xact_bf) {
  int i = blockIdx.x * blockDim.x + threadIdx.x;
  if (i >= ML_ * DI_) return;
  int row = i / DI_;               // b*L + t
  int d   = i - row * DI_;
  int t   = row & (L__ - 1);
  float acc = cb[d];
#pragma unroll
  for (int j = 0; j < KC_; ++j) {
    int tt = t - (KC_ - 1) + j;
    if (tt >= 0)
      acc += cw[d * KC_ + j] * xz[(size_t)(row - (KC_ - 1) + j) * (2 * DI_) + d];
  }
  float v = acc * sigmoidf_(acc);
  xact[i] = v;
  xact_bf[i] = f2bf(v);
}

// ---------------- fragment helpers (ISA 7.12.2 16-bit layouts) ----------------
__device__ __forceinline__ v16bf load_frag_global(const unsigned short* __restrict__ base,
                                                  int kb) {
  v8u raw;
#pragma unroll
  for (int v = 0; v < 8; ++v) {
    int kk = ((v >> 2) << 4) + kb + ((v & 3) << 1);
    raw[v] = *(const unsigned int*)(base + kk);
  }
  return __builtin_bit_cast(v16bf, raw);
}

#if HAVE_TDM
// Issue one TDM 2D tile load: 64 rows x KCH bf16 from W[row0..row0+63][k0..k0+127]
// into LDS at byte offset lds_off with 16B row padding (272B stride).
__device__ __forceinline__ void tdm_issue(const unsigned short* gsrc, unsigned lds_off,
                                          int K, int Np) {
  unsigned long long ga = (unsigned long long)(size_t)gsrc;
  u32x4 g0;
  g0[0] = 1u;                                            // count=1, user mode
  g0[1] = lds_off;                                       // lds_addr (bytes)
  g0[2] = (unsigned)(ga & 0xffffffffu);                  // global_addr[31:0]
  g0[3] = (unsigned)((ga >> 32) & 0x1ffffffu) | (2u << 30);  // addr[56:32] | type=2
  i32x8 g1;
  // data_size=2B, pad_enable, pad_interval=64 DW (code 5), pad_amount=4 DW (code 3)
  g1[0] = (1 << 16) | (1 << 20) | (5 << 22) | (3 << 25);
  g1[1] = (int)(((unsigned)K & 0xffffu) << 16);          // tensor_dim0 lo16 (bits 63:48)
  g1[2] = (int)(((unsigned)K >> 16) |                    // tensor_dim0 hi16
                (((unsigned)Np & 0xffffu) << 16));       // tensor_dim1 lo16
  g1[3] = (int)(((unsigned)Np >> 16) |                   // tensor_dim1 hi16
                ((unsigned)KCH << 16));                  // tile_dim0 = 128
  g1[4] = 64;                                            // tile_dim1 = 64, tile_dim2 = 0
  g1[5] = K;                                             // tensor_dim0_stride[31:0]
  g1[6] = 0;                                             // stride hi, dim1_stride lo
  g1[7] = 0;
  i32x4 z4 = {0, 0, 0, 0};
  i32x8 z8 = {0, 0, 0, 0, 0, 0, 0, 0};
  // 6-arg toolchain form: (g0, g1, g2, g3, g_extra, cpol)
  __builtin_amdgcn_tensor_load_to_lds(g0, g1, z4, z4, z8, 0);
}
#endif

// ---------------- WMMA GEMM (TDM-staged B): C[M,Nout] = A[M,K] * W[Np,K]^T ----
// Requires K % 128 == 0, Np % 64 == 0 (weights zero-padded). One wave / 16x64 strip.
__global__ __launch_bounds__(32) void wmma_gemm_tdm_k(
    const unsigned short* __restrict__ A, const unsigned short* __restrict__ W,
    float* __restrict__ C, int K, int Np, int Nout, int ldc) {
  __shared__ unsigned short sB[2][64 * ROWU];
  const int lane = threadIdx.x & 31;
  const int r    = lane & 15;
  const int half = lane >> 4;
  const int m0   = blockIdx.y << 4;
  const int n0   = blockIdx.x << 6;
  const int kb   = half * 8;

  const unsigned short* Arow  = A + (size_t)(m0 + r) * K;
  const unsigned short* Wbase = W + (size_t)n0 * K;
  v8f acc[4] = {};

#if HAVE_TDM
  unsigned short* sesc = &sB[0][0];          // escape LDS base into the fences
  const int nch = K >> 7;
  tdm_issue(Wbase, 0u, K, Np);
  for (int c = 0; c < nch; ++c) {
    if (c + 1 < nch) {
      tdm_issue(Wbase + (size_t)(c + 1) * KCH, (unsigned)(((c + 1) & 1) * LDSBUFB), K, Np);
      __builtin_amdgcn_s_wait_tensorcnt(1);   // chunk c resident, c+1 in flight
    } else {
      __builtin_amdgcn_s_wait_tensorcnt(0);
    }
    // Escaped-pointer fence: compiler must assume the TDM wrote sB, and may not
    // hoist the ds loads above the tensorcnt wait.
    asm volatile("" :: "v"(sesc) : "memory");
    const int buf = c & 1;
    if (c + 1 < nch) __builtin_prefetch(Arow + (c + 1) * KCH, 0, 0);
#pragma unroll
    for (int ks = 0; ks < 4; ++ks) {
      v16bf a = load_frag_global(Arow + c * KCH + ks * 32, kb);
#pragma unroll
      for (int j = 0; j < 4; ++j) {
        const int rr = (j << 4) + r;
        v8u braw;
#pragma unroll
        for (int v = 0; v < 8; ++v) {
          int kk = (ks << 5) + ((v >> 2) << 4) + kb + ((v & 3) << 1);
          braw[v] = *(const unsigned int*)&sB[buf][rr * ROWU + kk];
        }
        v16bf bb = __builtin_bit_cast(v16bf, braw);
        acc[j] = __builtin_amdgcn_wmma_f32_16x16x32_bf16(
            false, a, false, bb, (short)0, acc[j], false, false);
      }
    }
  }
#else
  for (int k0 = 0; k0 < K; k0 += 32) {
    v16bf a = load_frag_global(Arow + k0, kb);
    if (k0 + 32 < K) __builtin_prefetch(Arow + k0 + 32, 0, 0);
#pragma unroll
    for (int j = 0; j < 4; ++j) {
      v16bf bb = load_frag_global(W + (size_t)(n0 + (j << 4) + r) * K + k0, kb);
      acc[j] = __builtin_amdgcn_wmma_f32_16x16x32_bf16(
          false, a, false, bb, (short)0, acc[j], false, false);
    }
  }
  (void)Np;
#endif
#pragma unroll
  for (int j = 0; j < 4; ++j) {
    int n = n0 + (j << 4) + r;
    if (n < Nout) {
#pragma unroll
      for (int v = 0; v < 8; ++v) {
        int m = m0 + (half << 3) + v;
        C[(size_t)m * ldc + n] = acc[j][v];
      }
    }
  }
}

// ---------------- WMMA GEMM (direct, any K%32): optional bias+softplus -------
template <int EPI>
__global__ __launch_bounds__(32) void wmma_gemm_direct_k(
    const unsigned short* __restrict__ A, const unsigned short* __restrict__ W,
    float* __restrict__ C, const float* __restrict__ bias,
    int K, int Nout, int ldc) {
  const int lane = threadIdx.x & 31;
  const int r    = lane & 15;
  const int half = lane >> 4;
  const int m0   = blockIdx.y << 4;
  const int n0   = blockIdx.x << 6;
  const int kb   = half * 8;

  const unsigned short* Arow = A + (size_t)(m0 + r) * K;
  v8f acc[4] = {};
  for (int k0 = 0; k0 < K; k0 += 32) {
    v16bf a = load_frag_global(Arow + k0, kb);
#pragma unroll
    for (int j = 0; j < 4; ++j) {
      v16bf bb = load_frag_global(W + (size_t)(n0 + (j << 4) + r) * K + k0, kb);
      acc[j] = __builtin_amdgcn_wmma_f32_16x16x32_bf16(
          false, a, false, bb, (short)0, acc[j], false, false);
    }
  }
#pragma unroll
  for (int j = 0; j < 4; ++j) {
    int n = n0 + (j << 4) + r;
    if (n < Nout) {
#pragma unroll
      for (int v = 0; v < 8; ++v) {
        int m = m0 + (half << 3) + v;
        float val = acc[j][v];
        if (EPI == 1) {
          val += bias[n];
          val = (val > 20.f) ? val : log1pf(__expf(val));  // softplus
        }
        C[(size_t)m * ldc + n] = val;
      }
    }
  }
}

// ---------------- selective scan + skip + gate (L sequential) ----------------
__global__ __launch_bounds__(256) void scan_k(
    const float* __restrict__ delta, const float* __restrict__ xdbl,
    const float* __restrict__ xact,  const float* __restrict__ xz,
    const float* __restrict__ A_log, const float* __restrict__ Dp,
    unsigned short* __restrict__ ybf) {
  __shared__ float sB[2][NS_], sC[2][NS_];
  const int tid = threadIdx.x;
  const int b   = blockIdx.x >> 3;
  const int d   = ((blockIdx.x & 7) << 8) + tid;

  float Arow[NS_], h[NS_];
#pragma unroll
  for (int n = 0; n < NS_; ++n) { Arow[n] = -__expf(A_log[d * NS_ + n]); h[n] = 0.f; }
  const float Dd = Dp[d];

  for (int t = 0; t < L__; ++t) {
    size_t row = (size_t)b * L__ + t;
    int buf = t & 1;                           // ping-pong: 1 barrier/step
    if (tid < NS_)            sB[buf][tid]       = xdbl[row * XPN + RR_ + tid];
    else if (tid < 2 * NS_)   sC[buf][tid - NS_] = xdbl[row * XPN + RR_ + NS_ + (tid - NS_)];
    __syncthreads();
    float dt = delta[row * DI_ + d];
    float xt = xact[row * DI_ + d];
    float dx = dt * xt;
    float y = 0.f;
#pragma unroll
    for (int n = 0; n < NS_; ++n) {
      h[n] = __expf(dt * Arow[n]) * h[n] + dx * sB[buf][n];
      y += h[n] * sC[buf][n];
    }
    float z = xz[row * (2 * DI_) + DI_ + d];
    float out = (y + Dd * xt) * (z * sigmoidf_(z));
    ybf[row * DI_ + d] = f2bf(out);
  }
}

// ---------------- host side ----------------
extern "C" void kernel_launch(void* const* d_in, const int* in_sizes, int n_in,
                              void* d_out, int out_size, void* d_ws, size_t ws_size,
                              hipStream_t stream) {
  const float* hs        = (const float*)d_in[0];
  const float* in_proj_w = (const float*)d_in[1];
  const float* conv_w    = (const float*)d_in[2];
  const float* conv_b    = (const float*)d_in[3];
  const float* x_proj_w  = (const float*)d_in[4];
  const float* dt_proj_w = (const float*)d_in[5];
  const float* dt_proj_b = (const float*)d_in[6];
  const float* A_log     = (const float*)d_in[7];
  const float* D_param   = (const float*)d_in[8];
  const float* out_proj_w= (const float*)d_in[9];
  float* out = (float*)d_out;

  char* ws = (char*)d_ws;
  size_t off = 0;
  auto alloc = [&](size_t bytes) -> void* {
    void* p = ws + off;
    off += (bytes + 255) & ~(size_t)255;
    return p;
  };
  unsigned short* hs_bf    = (unsigned short*)alloc((size_t)ML_ * DM_ * 2);
  unsigned short* w_in_bf  = (unsigned short*)alloc((size_t)2 * DI_ * DM_ * 2);
  unsigned short* w_x_bf   = (unsigned short*)alloc((size_t)XPNP * DI_ * 2);  // padded rows
  unsigned short* w_dt_bf  = (unsigned short*)alloc((size_t)DI_ * RR_ * 2);
  unsigned short* w_out_bf = (unsigned short*)alloc((size_t)DM_ * DI_ * 2);
  float*          xz       = (float*)alloc((size_t)ML_ * 2 * DI_ * 4);
  float*          xact     = (float*)alloc((size_t)ML_ * DI_ * 4);
  unsigned short* xact_bf  = (unsigned short*)alloc((size_t)ML_ * DI_ * 2);
  float*          xdbl     = (float*)alloc((size_t)ML_ * XPN * 4);
  unsigned short* dtlow_bf = (unsigned short*)alloc((size_t)ML_ * RR_ * 2);
  float*          delta    = (float*)alloc((size_t)ML_ * DI_ * 4);
  unsigned short* ybf      = (unsigned short*)alloc((size_t)ML_ * DI_ * 2);

  const int T = 256;
  int n;
  n = ML_ * DM_;       cast_f32_bf16_k<<<(n + T - 1) / T, T, 0, stream>>>(hs, hs_bf, n);
  n = 2 * DI_ * DM_;   cast_f32_bf16_k<<<(n + T - 1) / T, T, 0, stream>>>(in_proj_w, w_in_bf, n);
  n = XPNP * DI_;      cast_pad_bf16_k<<<(n + T - 1) / T, T, 0, stream>>>(x_proj_w, w_x_bf, XPN, XPNP, DI_);
  n = DI_ * RR_;       cast_f32_bf16_k<<<(n + T - 1) / T, T, 0, stream>>>(dt_proj_w, w_dt_bf, n);
  n = DM_ * DI_;       cast_f32_bf16_k<<<(n + T - 1) / T, T, 0, stream>>>(out_proj_w, w_out_bf, n);

  // in_proj: (4096 x 1024) x (1024 x 4096) -> xz, ldc = 4096  [TDM-staged]
  {
    dim3 g((2 * DI_) / 64, ML_ / 16);
    wmma_gemm_tdm_k<<<g, 32, 0, stream>>>(hs_bf, w_in_bf, xz, DM_, 2 * DI_, 2 * DI_, 2 * DI_);
  }
  // depthwise conv + SiLU
  n = ML_ * DI_;
  conv_silu_k<<<(n + T - 1) / T, T, 0, stream>>>(xz, conv_w, conv_b, xact, xact_bf);

  // x_proj: (4096 x 2048) x (2048 x 96) -> xdbl (weights padded to 128 rows) [TDM]
  {
    dim3 g(XPNP / 64, ML_ / 16);
    wmma_gemm_tdm_k<<<g, 32, 0, stream>>>(xact_bf, w_x_bf, xdbl, DI_, XPNP, XPN, XPN);
  }
  // dt_low slice -> bf16
  n = ML_ * RR_;
  slice_bf16_k<<<(n + T - 1) / T, T, 0, stream>>>(xdbl, dtlow_bf, ML_, RR_, XPN);

  // dt_proj + bias + softplus: (4096 x 64) x (64 x 2048) -> delta  [direct, K=64]
  {
    dim3 g(DI_ / 64, ML_ / 16);
    wmma_gemm_direct_k<1><<<g, 32, 0, stream>>>(dtlow_bf, w_dt_bf, delta, dt_proj_b, RR_, DI_, DI_);
  }
  // selective scan + D*x skip + SiLU(z) gate -> ybf (bf16)
  scan_k<<<B__ * (DI_ / 256), 256, 0, stream>>>(delta, xdbl, xact, xz, A_log, D_param, ybf);

  // out_proj: (4096 x 2048) x (2048 x 1024) -> out  [TDM]
  {
    dim3 g(DM_ / 64, ML_ / 16);
    wmma_gemm_tdm_k<<<g, 32, 0, stream>>>(ybf, w_out_bf, out, DI_, DM_, DM_, DM_);
  }
}
